// LiTranEnc_49692771614982
// MI455X (gfx1250) — matrix operand order repaired and checked by reading, not compiled, not run on hardware
//
#include <hip/hip_runtime.h>
#include <hip/hip_bf16.h>

// ---- problem constants ----
constexpr int kD   = 512;
constexpr int kH   = 8;
constexpr int kHD  = 64;
constexpr int kFFD = 2048;
constexpr int kNL  = 6;
constexpr int kB   = 16;
constexpr int kL   = 2048;
constexpr int kM   = kB * kL;      // 32768 rows
#define LN_EPS 1e-5f

typedef __bf16 bf16_t;
typedef __bf16 bf16x8 __attribute__((ext_vector_type(8)));
typedef __bf16 v16bf  __attribute__((ext_vector_type(16)));
typedef float  v8f    __attribute__((ext_vector_type(8)));

__device__ __forceinline__ bf16_t f2bf(float f) {
  union { float f; unsigned u; } uf; uf.f = f;
  unsigned r = uf.u + 0x7FFFu + ((uf.u >> 16) & 1u);   // round-to-nearest-even
  unsigned short hs = (unsigned short)(r >> 16);
  bf16_t b; __builtin_memcpy(&b, &hs, 2); return b;
}

// async memory -> LDS copy, 16 bytes per lane, tracked by ASYNCcnt
__device__ __forceinline__ void async_b128(unsigned lds_off, const void* gptr) {
  asm volatile("global_load_async_to_lds_b128 %0, %1, off"
               :: "v"(lds_off), "v"((unsigned long long)(size_t)gptr)
               : "memory");
}

// load a 16x16 bf16 WMMA fragment: two contiguous bf16x8 pieces, `delta`
// elements apart (A frag: delta=16; B frag: delta=16), per ISA VGPR layout
__device__ __forceinline__ v16bf ld_frag(const bf16_t* base, int delta) {
  bf16x8 lo = *(const bf16x8*)base;
  bf16x8 hi = *(const bf16x8*)(base + delta);
  v16bf r;
#pragma unroll
  for (int i = 0; i < 8; ++i) { r[i] = lo[i]; r[i + 8] = hi[i]; }
  return r;
}

__device__ __forceinline__ v8f wmma_bf16(v16bf a, v16bf b, v8f c) {
  return __builtin_amdgcn_wmma_f32_16x16x32_bf16(
      false, a, false, b, (short)0, c, false, false);
}

// ---------------------------------------------------------------------------
// conversion kernels
// ---------------------------------------------------------------------------
__global__ void cvt_f32_bf16_kernel(const float* __restrict__ in,
                                    bf16_t* __restrict__ out, int n) {
  int i = blockIdx.x * 256 + threadIdx.x;
  if (i < n) out[i] = f2bf(in[i]);
}

// in: [g][o*kD+d][3] fp32  ->  out: [g][t][o*kD+d] bf16
__global__ void cvt_convw_kernel(const float* __restrict__ in,
                                 bf16_t* __restrict__ out, int nmat) {
  int i = blockIdx.x * 256 + threadIdx.x;
  int tot = nmat * kD * kD;
  if (i >= tot) return;
  int g = i / (kD * kD), od = i % (kD * kD);
  const float* src = in + (size_t)g * kD * kD * 3 + (size_t)od * 3;
  bf16_t* dst = out + (size_t)g * 3 * kD * kD + od;
  dst[0]           = f2bf(src[0]);
  dst[kD * kD]     = f2bf(src[1]);
  dst[2 * kD * kD] = f2bf(src[2]);
}

#define LDSTRIDE 40  // 32 + 8 pad (80B row stride, keeps 16B alignment)

// ---------------------------------------------------------------------------
// GEMM: C(kM x N) = act(A(kM x K,bf16) * Bw(N x K,bf16)^T + bias)
// block tile 128(M) x 64(N), BK=32, 8 waves; each wave owns a 32x32 tile
// (2x2 WMMA 16x16x32). LDS triple-buffered with async-to-LDS copies:
// one barrier per K-step.  act: 0 none, 1 relu, 2 sigmoid.
// ---------------------------------------------------------------------------
template <int N, int K>
__global__ __launch_bounds__(256)
void gemm_bf16_kernel(const bf16_t* __restrict__ A, const bf16_t* __restrict__ Bw,
                      const float* __restrict__ bias,
                      float* __restrict__ Cf, bf16_t* __restrict__ Cb, int act) {
  __shared__ __attribute__((aligned(16))) bf16_t As[3][128][LDSTRIDE];
  __shared__ __attribute__((aligned(16))) bf16_t Bs[3][64][LDSTRIDE];
  const int tid  = threadIdx.x;
  const int lane = tid & 31;
  const int wave = tid >> 5;
  const int wr   = wave & 3;     // 32-row slab within 128
  const int wc   = wave >> 2;    // 32-col slab within 64
  const int half = lane >> 4;
  const int l15  = lane & 15;
  const int blockN = blockIdx.x * 64;
  const int blockM = blockIdx.y * 128;

  // staging map: 3 async b128 per thread per K-step
  const int ar0 = tid >> 2;           // A rows 0..63
  const int ar1 = ar0 + 64;           // A rows 64..127
  const int ac  = (tid & 3) * 8;      // col 0,8,16,24
  const int br  = tid >> 2;           // B rows 0..63
  const bf16_t* Arow0 = A + (size_t)(blockM + ar0) * K + ac;
  const bf16_t* Arow1 = A + (size_t)(blockM + ar1) * K + ac;
  const bf16_t* Brow  = Bw + (size_t)(blockN + br) * K + ac;

  auto issue = [&](int buf, int k1) {
    async_b128((unsigned)(size_t)&As[buf][ar0][ac], Arow0 + k1);
    async_b128((unsigned)(size_t)&As[buf][ar1][ac], Arow1 + k1);
    async_b128((unsigned)(size_t)&Bs[buf][br][ac],  Brow + k1);
  };

  v8f acc00 = {}, acc01 = {}, acc10 = {}, acc11 = {};
  issue(0, 0);
  constexpr int S = K / 32;
  int buf = 0;
  for (int s = 0; s < S; ++s) {
    const int nx = (buf == 2) ? 0 : buf + 1;
    if (s + 1 < S) {
      issue(nx, (s + 1) * 32);
      asm volatile("s_wait_asynccnt 0x3" ::: "memory");  // current 3 landed
    } else {
      asm volatile("s_wait_asynccnt 0x0" ::: "memory");
    }
    __syncthreads();   // single barrier/step: triple buffer makes WAR safe

    v16bf a0 = ld_frag(&As[buf][wr * 32 + l15][half * 8], 16);
    v16bf a1 = ld_frag(&As[buf][wr * 32 + 16 + l15][half * 8], 16);
    v16bf b0 = ld_frag(&Bs[buf][wc * 32 + l15][half * 8], 16);
    v16bf b1 = ld_frag(&Bs[buf][wc * 32 + 16 + l15][half * 8], 16);
    acc00 = wmma_bf16(a0, b0, acc00);
    acc01 = wmma_bf16(a0, b1, acc01);
    acc10 = wmma_bf16(a1, b0, acc10);
    acc11 = wmma_bf16(a1, b1, acc11);
    buf = nx;
  }

  const v8f accs[4] = {acc00, acc01, acc10, acc11};
#pragma unroll
  for (int t = 0; t < 4; ++t) {
    int mb = wr * 32 + (t >> 1) * 16;
    int nb = wc * 32 + (t & 1) * 16;
    int n = blockN + nb + l15;
    float bsv = bias ? bias[n] : 0.f;
#pragma unroll
    for (int r = 0; r < 8; ++r) {
      int m = blockM + mb + half * 8 + r;
      float v = accs[t][r] + bsv;
      if (act == 1)      v = fmaxf(v, 0.f);
      else if (act == 2) v = 1.f / (1.f + __expf(-v));
      if (Cf) Cf[(size_t)m * N + n] = v;
      if (Cb) Cb[(size_t)m * N + n] = f2bf(v);
    }
  }
}

// ---------------------------------------------------------------------------
// conv1d(k=3, pad=1) via WMMA.  Per K-step we stage ONE 130-row A slab
// (rows blockM-1 .. blockM+128) and the 3 taps' B tiles, then run 3x4 WMMAs
// reusing the slab with a +t row shift.  Boundary rows (tile row 0/129 at a
// sequence edge) are pre-zeroed and skipped during staging; skipped/overflow
// lanes redirect to a dummy LDS slot so every wave issues exactly 6 async
// ops per step (uniform ASYNCcnt).  Triple-buffered, 1 barrier per step.
// X: (kM, kD) bf16;  Wt: [3][kD][kD] bf16 ([t][o][d]);  Y: (kM, kD) f32
// ---------------------------------------------------------------------------
__global__ __launch_bounds__(256)
void conv3_bf16_kernel(const bf16_t* __restrict__ X, const bf16_t* __restrict__ Wt,
                       const float* __restrict__ bias, float* __restrict__ Y) {
  __shared__ __attribute__((aligned(16))) bf16_t As[3][130][LDSTRIDE];
  __shared__ __attribute__((aligned(16))) bf16_t Bs[3][3][64][LDSTRIDE];
  __shared__ __attribute__((aligned(16))) bf16_t Dummy[8];
  const int tid  = threadIdx.x;
  const int lane = tid & 31;
  const int wave = tid >> 5;
  const int wr   = wave & 3;
  const int wc   = wave >> 2;
  const int half = lane >> 4;
  const int l15  = lane & 15;
  const int blockN = blockIdx.x * 64;
  const int blockM = blockIdx.y * 128;

  // at most one staged row is outside its sequence (tiles are 128-aligned)
  const int inv0 = ((blockM & (kL - 1)) == 0) ? 0 : -1;          // l == -1
  const int inv1 = (((blockM + 128) & (kL - 1)) == 0) ? 129 : -1; // l == kL
  // pre-zero both candidate boundary rows in all 3 buffers
  if (tid < 24) {
    int bf = tid >> 3, c = tid & 7;
    int row = (c >> 2) ? 129 : 0;
    int col = (c & 3) * 8;
    *(uint4*)(&As[bf][row][col]) = make_uint4(0u, 0u, 0u, 0u);
  }
  __syncthreads();

  auto issueA = [&](int buf, int k0) {   // 130 rows = 520 chunks, 3/thread
#pragma unroll
    for (int i = 0; i < 3; ++i) {
      int idx = tid + i * 256;
      int row = idx >> 2;
      int col = (idx & 3) * 8;
      bool real = (idx < 520) && (row != inv0) && (row != inv1);
      int gr = blockM - 1 + row;
      gr = (gr < 0) ? 0 : ((gr >= kM) ? kM - 1 : gr);
      unsigned laddr = real ? (unsigned)(size_t)&As[buf][row][col]
                            : (unsigned)(size_t)&Dummy[0];
      const bf16_t* g = real ? (X + (size_t)gr * kD + k0 + col) : X;
      async_b128(laddr, g);              // per-lane address select, no branch
    }
  };
  auto issueB = [&](int buf, int k0) {   // 3 taps x (64x32), 3/thread
    int row = tid >> 2, col = (tid & 3) * 8;
#pragma unroll
    for (int t = 0; t < 3; ++t)
      async_b128((unsigned)(size_t)&Bs[buf][t][row][col],
                 Wt + ((size_t)t * kD + (size_t)(blockN + row)) * kD + k0 + col);
  };

  v8f acc00 = {}, acc01 = {}, acc10 = {}, acc11 = {};
  issueA(0, 0);
  issueB(0, 0);
  constexpr int S = kD / 32;
  int buf = 0;
  for (int s = 0; s < S; ++s) {
    const int nx = (buf == 2) ? 0 : buf + 1;
    if (s + 1 < S) {
      issueA(nx, (s + 1) * 32);
      issueB(nx, (s + 1) * 32);
      asm volatile("s_wait_asynccnt 0x6" ::: "memory");  // current 6 landed
    } else {
      asm volatile("s_wait_asynccnt 0x0" ::: "memory");
    }
    __syncthreads();

#pragma unroll
    for (int t = 0; t < 3; ++t) {        // tap t reads A rows shifted by +t
      v16bf a0 = ld_frag(&As[buf][wr * 32 + l15 + t][half * 8], 16);
      v16bf a1 = ld_frag(&As[buf][wr * 32 + 16 + l15 + t][half * 8], 16);
      v16bf b0 = ld_frag(&Bs[buf][t][wc * 32 + l15][half * 8], 16);
      v16bf b1 = ld_frag(&Bs[buf][t][wc * 32 + 16 + l15][half * 8], 16);
      acc00 = wmma_bf16(a0, b0, acc00);
      acc01 = wmma_bf16(a0, b1, acc01);
      acc10 = wmma_bf16(a1, b0, acc10);
      acc11 = wmma_bf16(a1, b1, acc11);
    }
    buf = nx;
  }

  const v8f accs[4] = {acc00, acc01, acc10, acc11};
#pragma unroll
  for (int t = 0; t < 4; ++t) {
    int mb = wr * 32 + (t >> 1) * 16;
    int nb = wc * 32 + (t & 1) * 16;
    int n = blockN + nb + l15;
    float bsv = bias[n];
#pragma unroll
    for (int r = 0; r < 8; ++r) {
      int m = blockM + mb + half * 8 + r;
      Y[(size_t)m * kD + n] = accs[t][r] + bsv;
    }
  }
}

// ---------------------------------------------------------------------------
// global max over L of k,v per (b, h*HD+d)
// ---------------------------------------------------------------------------
__global__ void kvmax_kernel(const float* __restrict__ K, const float* __restrict__ V,
                             float* __restrict__ kg, float* __restrict__ vg) {
  int idx = blockIdx.x * 256 + threadIdx.x;   // < kB*kD
  if (idx >= kB * kD) return;
  int b = idx / kD, hd = idx % kD;
  const float* kp = K + (size_t)b * kL * kD + hd;
  const float* vp = V + (size_t)b * kL * kD + hd;
  float mk = -3.4e38f, mv = -3.4e38f;
  for (int l = 0; l < kL; ++l) {
    mk = fmaxf(mk, kp[(size_t)l * kD]);
    mv = fmaxf(mv, vp[(size_t)l * kD]);
  }
  kg[idx] = mk; vg[idx] = mv;
}

// ---------------------------------------------------------------------------
// 2-way (local/global) softmax attention combine; one wave per (b,l,h)
// ---------------------------------------------------------------------------
__global__ __launch_bounds__(256)
void attn_kernel(const float* __restrict__ Q, const float* __restrict__ Kk,
                 const float* __restrict__ V, const float* __restrict__ kg,
                 const float* __restrict__ vg, bf16_t* __restrict__ Ob) {
  int wave = threadIdx.x >> 5, lane = threadIdx.x & 31;
  int row = blockIdx.x * 8 + wave;    // over kM*kH
  int bl = row / kH;                  // b*kL + l
  int h  = row % kH;
  size_t base = (size_t)bl * kD + (size_t)h * kHD;
  float q0 = Q[base + lane],  q1 = Q[base + 32 + lane];
  float k0 = Kk[base + lane], k1 = Kk[base + 32 + lane];
  float v0 = V[base + lane],  v1 = V[base + 32 + lane];
  int b = bl / kL;
  size_t gbase = (size_t)b * kD + (size_t)h * kHD;
  float g0 = kg[gbase + lane], g1 = kg[gbase + 32 + lane];
  float w0 = vg[gbase + lane], w1 = vg[gbase + 32 + lane];
  float sl = q0 * k0 + q1 * k1;
  float sg = q0 * g0 + q1 * g1;
#pragma unroll
  for (int off = 16; off; off >>= 1) {
    sl += __shfl_xor(sl, off, 32);
    sg += __shfl_xor(sg, off, 32);
  }
  const float scale = 0.04419417382415922f;  // 1/sqrt(512)
  sl *= scale; sg *= scale;
  float mx = fmaxf(sl, sg);
  float e0 = __expf(sl - mx), e1 = __expf(sg - mx);
  float inv = 1.f / (e0 + e1);
  float a0 = e0 * inv, a1 = e1 * inv;
  Ob[base + lane]      = f2bf(a0 * v0 + a1 * w0);
  Ob[base + 32 + lane] = f2bf(a0 * v1 + a1 * w1);
}

// ---------------------------------------------------------------------------
// x = LayerNorm(x + add * (gate?) * mask); also emit bf16 copy
// one block (256 thr) per row of 512
// ---------------------------------------------------------------------------
__global__ __launch_bounds__(256)
void res_ln_kernel(float* __restrict__ x, bf16_t* __restrict__ xb,
                   const float* __restrict__ add, const float* __restrict__ gate,
                   const int* __restrict__ mask,
                   const float* __restrict__ g, const float* __restrict__ b) {
  int row = blockIdx.x;
  int tid = threadIdx.x;
  float m = (float)mask[row];
  size_t base = (size_t)row * kD;
  float gv0 = gate ? gate[base + tid] : 1.f;
  float gv1 = gate ? gate[base + tid + 256] : 1.f;
  float t0 = x[base + tid]       + add[base + tid]       * gv0 * m;
  float t1 = x[base + tid + 256] + add[base + tid + 256] * gv1 * m;
  float s = t0 + t1, s2 = t0 * t0 + t1 * t1;
#pragma unroll
  for (int off = 16; off; off >>= 1) {
    s  += __shfl_xor(s, off, 32);
    s2 += __shfl_xor(s2, off, 32);
  }
  __shared__ float red[16];
  int lane = tid & 31, wave = tid >> 5;
  if (lane == 0) { red[wave] = s; red[wave + 8] = s2; }
  __syncthreads();
  float mean = 0.f, msq = 0.f;
#pragma unroll
  for (int w = 0; w < 8; ++w) { mean += red[w]; msq += red[w + 8]; }
  mean *= (1.f / kD);
  float var = msq * (1.f / kD) - mean * mean;
  float inv = rsqrtf(var + LN_EPS);
  float y0 = (t0 - mean) * inv * g[tid]       + b[tid];
  float y1 = (t1 - mean) * inv * g[tid + 256] + b[tid + 256];
  x[base + tid] = y0;        x[base + tid + 256] = y1;
  xb[base + tid] = f2bf(y0); xb[base + tid + 256] = f2bf(y1);
}

// ---------------------------------------------------------------------------
// host launcher
// ---------------------------------------------------------------------------
extern "C" void kernel_launch(void* const* d_in, const int* in_sizes, int n_in,
                              void* d_out, int out_size, void* d_ws, size_t ws_size,
                              hipStream_t stream) {
  (void)in_sizes; (void)n_in; (void)out_size; (void)ws_size;
  const float* in_x   = (const float*)d_in[0];
  const int*   in_msk = (const int*)d_in[1];
  const float* w_qkv  = (const float*)d_in[2];
  const float* b_qkv  = (const float*)d_in[3];
  const float* w_gate = (const float*)d_in[4];
  const float* b_gate = (const float*)d_in[5];
  const float* w_out  = (const float*)d_in[6];
  const float* b_out  = (const float*)d_in[7];
  const float* ln1_g  = (const float*)d_in[8];
  const float* ln1_b  = (const float*)d_in[9];
  const float* w_ff1  = (const float*)d_in[10];
  const float* b_ff1  = (const float*)d_in[11];
  const float* w_ff2  = (const float*)d_in[12];
  const float* b_ff2  = (const float*)d_in[13];
  const float* ln2_g  = (const float*)d_in[14];
  const float* ln2_b  = (const float*)d_in[15];

  char* wsb = (char*)d_ws;
  size_t off = 0;
  auto alloc = [&](size_t bytes) -> char* {
    size_t p = (off + 255) & ~(size_t)255;
    off = p + bytes;
    return wsb + p;
  };
  const size_t NE = (size_t)kM * kD;
  const size_t NF = (size_t)kM * kFFD;
  float*  x    = (float*)alloc(NE * 4);
  bf16_t* xb   = (bf16_t*)alloc(NE * 2);
  float*  gate = (float*)alloc(NE * 4);
  float*  q    = (float*)alloc(NE * 4);
  float*  k    = (float*)alloc(NE * 4);
  float*  v    = (float*)alloc(NE * 4);
  float*  kg   = (float*)alloc((size_t)kB * kD * 4);
  float*  vg   = (float*)alloc((size_t)kB * kD * 4);
  bf16_t* ob   = (bf16_t*)alloc(NE * 2);
  float*  oc   = (float*)alloc(NE * 4);
  bf16_t* hbuf = (bf16_t*)alloc(NF * 2);
  float*  fbuf = (float*)alloc(NE * 4);
  bf16_t* wgate_bf = (bf16_t*)alloc((size_t)kNL * kD * kD * 2);
  bf16_t* wqkv_bf  = (bf16_t*)alloc((size_t)kNL * 3 * 3 * kD * kD * 2);
  bf16_t* wout_bf  = (bf16_t*)alloc((size_t)kNL * 3 * kD * kD * 2);
  bf16_t* wff1_bf  = (bf16_t*)alloc((size_t)kNL * kFFD * kD * 2);
  bf16_t* wff2_bf  = (bf16_t*)alloc((size_t)kNL * kD * kFFD * 2);

  hipMemcpyAsync(x, in_x, NE * 4, hipMemcpyDeviceToDevice, stream);
  { int n = (int)NE;           cvt_f32_bf16_kernel<<<(n + 255) / 256, 256, 0, stream>>>(in_x, xb, n); }
  { int n = kNL * kD * kD;     cvt_f32_bf16_kernel<<<(n + 255) / 256, 256, 0, stream>>>(w_gate, wgate_bf, n); }
  { int n = kNL * 3 * kD * kD; cvt_convw_kernel<<<(n + 255) / 256, 256, 0, stream>>>(w_qkv, wqkv_bf, kNL * 3); }
  { int n = kNL * kD * kD;     cvt_convw_kernel<<<(n + 255) / 256, 256, 0, stream>>>(w_out, wout_bf, kNL); }
  { int n = kNL * kFFD * kD;   cvt_f32_bf16_kernel<<<(n + 255) / 256, 256, 0, stream>>>(w_ff1, wff1_bf, n); }
  { int n = kNL * kD * kFFD;   cvt_f32_bf16_kernel<<<(n + 255) / 256, 256, 0, stream>>>(w_ff2, wff2_bf, n); }

  dim3 blk(256);
  dim3 gD(kD / 64, kM / 128);      // N=512 GEMMs / convs
  dim3 gF(kFFD / 64, kM / 128);    // N=2048 (FF1)
  for (int i = 0; i < kNL; ++i) {
    // gate = sigmoid(x @ w_gate^T + b)
    gemm_bf16_kernel<kD, kD><<<gD, blk, 0, stream>>>(
        xb, wgate_bf + (size_t)i * kD * kD, b_gate + (size_t)i * kD,
        gate, nullptr, 2);
    // q, k, v = conv3(x)
    conv3_bf16_kernel<<<gD, blk, 0, stream>>>(
        xb, wqkv_bf + (size_t)(i * 3 + 0) * 3 * kD * kD, b_qkv + (size_t)(i * 3 + 0) * kD, q);
    conv3_bf16_kernel<<<gD, blk, 0, stream>>>(
        xb, wqkv_bf + (size_t)(i * 3 + 1) * 3 * kD * kD, b_qkv + (size_t)(i * 3 + 1) * kD, k);
    conv3_bf16_kernel<<<gD, blk, 0, stream>>>(
        xb, wqkv_bf + (size_t)(i * 3 + 2) * 3 * kD * kD, b_qkv + (size_t)(i * 3 + 2) * kD, v);
    kvmax_kernel<<<dim3((kB * kD + 255) / 256), blk, 0, stream>>>(k, v, kg, vg);
    attn_kernel<<<dim3(kM * kH / 8), blk, 0, stream>>>(q, k, v, kg, vg, ob);
    // out conv
    conv3_bf16_kernel<<<gD, blk, 0, stream>>>(
        ob, wout_bf + (size_t)i * 3 * kD * kD, b_out + (size_t)i * kD, oc);
    // x = LN1(x + oc*gate*mask)
    res_ln_kernel<<<dim3(kM), blk, 0, stream>>>(
        x, xb, oc, gate, in_msk, ln1_g + (size_t)i * kD, ln1_b + (size_t)i * kD);
    // h = relu(x @ w_ff1^T + b) (bf16 out), f = h @ w_ff2^T + b
    gemm_bf16_kernel<kFFD, kD><<<gF, blk, 0, stream>>>(
        xb, wff1_bf + (size_t)i * kFFD * kD, b_ff1 + (size_t)i * kFFD,
        nullptr, hbuf, 1);
    gemm_bf16_kernel<kD, kFFD><<<gD, blk, 0, stream>>>(
        hbuf, wff2_bf + (size_t)i * kD * kFFD, b_ff2 + (size_t)i * kD,
        fbuf, nullptr, 0);
    // x = LN2(x + f*mask)
    res_ln_kernel<<<dim3(kM), blk, 0, stream>>>(
        x, xb, fbuf, nullptr, in_msk, ln2_g + (size_t)i * kD, ln2_b + (size_t)i * kD);
  }
  hipMemcpyAsync(d_out, x, NE * 4, hipMemcpyDeviceToDevice, stream);
}
